// EncoderLayer_66159676228247
// MI455X (gfx1250) — matrix operand order repaired and checked by reading, not compiled, not run on hardware
//
#include <hip/hip_runtime.h>

// ---------------------------------------------------------------------------
// CDNA5 (gfx1250) types & helpers
// ---------------------------------------------------------------------------
typedef __attribute__((ext_vector_type(16))) __bf16 v16bf;
typedef __attribute__((ext_vector_type(8)))  float  v8f;

union FragU {
    uint4          q[2];
    unsigned short h[16];
    v16bf          v;
};
union HalfU {           // 8 bf16 = one 16B store
    uint4          q;
    unsigned short h[8];
};

__device__ __forceinline__ unsigned short f2bf(float f) {
    return __builtin_bit_cast(unsigned short, (__bf16)f);   // native v_cvt, RNE
}

// raw v_exp_f32 (base-2), no libm range guards: ideal for softmax (args <= 0)
__device__ __forceinline__ float fast_exp2(float x) {
    return __builtin_amdgcn_exp2f(x);
}

// two contiguous 16B runs -> one 16-element bf16 fragment
__device__ __forceinline__ v16bf ld_frag_pair(const unsigned short* p0,
                                              const unsigned short* p1) {
    FragU u;
    u.q[0] = *reinterpret_cast<const uint4*>(p0);
    u.q[1] = *reinterpret_cast<const uint4*>(p1);
    return u.v;
}

__device__ __forceinline__ v8f wmma_bf16(v16bf a, v16bf b, v8f c) {
    return __builtin_amdgcn_wmma_f32_16x16x32_bf16(
        false, a, false, b, (short)0, c, false, false);
}

// ---- CDNA5 async global->LDS copy path (ASYNCcnt) --------------------------
// generic pointer to a __shared__ object: low 32 bits are the LDS offset
// (ISA 10.2 aperture rule)
__device__ __forceinline__ unsigned lds_addr(const void* p) {
    return (unsigned)(size_t)p;
}
// GLOBAL_LOAD_ASYNC_TO_LDS_B128, GV mode (CDNA5 ISA)
__device__ __forceinline__ void async_copy_b128(unsigned ldsOff, const void* gaddr) {
    asm volatile("global_load_async_to_lds_b128 %0, %1, off"
                 :: "v"(ldsOff), "v"(gaddr) : "memory");
}
__device__ __forceinline__ void wait_async0() {
    asm volatile("s_wait_asynccnt 0x0" ::: "memory");
}

// ---------------------------------------------------------------------------
// f32 -> bf16 conversion
// ---------------------------------------------------------------------------
__global__ __launch_bounds__(256) void cvt_bf16_kernel(const float* __restrict__ in,
                                                       unsigned short* __restrict__ out,
                                                       size_t n) {
    size_t i = (size_t)blockIdx.x * 256 + threadIdx.x;
    size_t stride = (size_t)gridDim.x * 256;
    for (; i < n; i += stride) out[i] = f2bf(in[i]);
}

// ---------------------------------------------------------------------------
// Pack weight W[K][N] (f32 row-major) into per-lane WMMA B-fragment layout:
// packed[((kt*(N/16)+nt)*32 + lane)*16 + i] = bf16(W[kt*32 + 16*(lane>>4) + i][nt*16 + (lane&15)])
// -> a lane's fragment is one contiguous 32B load; a wave's loads are 1KB contiguous.
// ---------------------------------------------------------------------------
__global__ __launch_bounds__(256) void packw_kernel(const float* __restrict__ W,
                                                    unsigned short* __restrict__ out,
                                                    int K, int N) {
    size_t total = (size_t)K * N;
    size_t t = (size_t)blockIdx.x * 256 + threadIdx.x;
    if (t >= total) return;
    int    i    = (int)(t & 15);
    int    lane = (int)((t >> 4) & 31);
    size_t tile = t >> 9;
    int    ntiles = N >> 4;
    int    nt = (int)(tile % ntiles);
    int    kt = (int)(tile / ntiles);
    int    k  = kt * 32 + ((lane >> 4) << 4) + i;
    int    n  = nt * 16 + (lane & 15);
    out[t] = f2bf(W[(size_t)k * N + n]);
}

// ---------------------------------------------------------------------------
// Tiled GEMM: C[M,N] = A_bf16[M,K] @ Bpacked_bf16[K,N] + bias (+resid) (+relu)
// 256 threads = 8 waves as 2(M) x 4(N); block tile 128 x 128; BK = 32.
// Each wave: 64x32 C tile = 4x2 v_wmma_f32_16x16x32_bf16 accumulators.
// A tile staged with async global->LDS DMA, double-buffered: issue copy for
// k+1, run WMMAs on k, s_wait_asynccnt 0, one barrier per k-step.
// LAYOUT 0: row-major out. 1: bf16 out -> [B,H,S,64]. 2: bf16 out -> [B,H,64,S].
// ---------------------------------------------------------------------------
template<int LAYOUT, bool RELU, bool RESID, bool OUTF, bool OUTB>
__global__ __launch_bounds__(256) void gemm_kernel(const unsigned short* __restrict__ A,
                                                   const unsigned short* __restrict__ Bp,
                                                   const float* __restrict__ bias,
                                                   const float* __restrict__ resid,
                                                   float* __restrict__ outF,
                                                   unsigned short* __restrict__ outB,
                                                   int M, int N, int K) {
    __shared__ alignas(16) unsigned short Alds[2][128 * 40];   // 80B pitch: conflict-free frag reads

    const int tid  = threadIdx.x;
    const int lane = tid & 31;
    const int wid  = tid >> 5;
    const int hf   = lane >> 4;
    const int ln   = lane & 15;
    const int wn   = wid & 3;        // wave N index (0..3)
    const int wm   = wid >> 2;       // wave M index (0..1)
    const int blockM = blockIdx.y * 128;
    const int blockN = blockIdx.x * 128;

    // A staging: 2 threads per row, 32B (2 x b128 async copies) each
    const int rowS = tid >> 1;
    const int segS = tid & 1;
    const unsigned short* aSrc = A + (size_t)(blockM + rowS) * K + segS * 16;
    const unsigned ldsS0 = lds_addr(&Alds[0][rowS * 40 + segS * 16]);
    const unsigned ldsS1 = lds_addr(&Alds[1][rowS * 40 + segS * 16]);

    v8f acc[4][2];
#pragma unroll
    for (int mi = 0; mi < 4; ++mi)
#pragma unroll
        for (int ni = 0; ni < 2; ++ni)
#pragma unroll
            for (int r = 0; r < 8; ++r) acc[mi][ni][r] = 0.0f;

    const int ksteps = K >> 5;
    const int ntileN = N >> 4;

    // preload k-tile 0 via async DMA
    async_copy_b128(ldsS0,      aSrc);
    async_copy_b128(ldsS0 + 16, aSrc + 8);
    wait_async0();
    __syncthreads();

    for (int kt = 0; kt < ksteps; ++kt) {
        const int cur = kt & 1;
        const bool more = (kt + 1) < ksteps;
        if (more) {
            const unsigned short* nsrc = aSrc + (size_t)(kt + 1) * 32;
            const unsigned nl = cur ? ldsS0 : ldsS1;
            async_copy_b128(nl,      nsrc);
            async_copy_b128(nl + 16, nsrc + 8);
            if (kt + 2 < ksteps)
                __builtin_prefetch(aSrc + (size_t)(kt + 2) * 32, 0, 0);  // global_prefetch_b8
        }

        // A fragments (rows = wm*64 + mi*16 + ln; runs at {16h, 16h+8})
        v16bf af[4];
#pragma unroll
        for (int mi = 0; mi < 4; ++mi) {
            const int r = (wm * 64 + mi * 16 + ln) * 40;
            af[mi] = ld_frag_pair(&Alds[cur][r + 16 * hf], &Alds[cur][r + 16 * hf + 8]);
        }
        // B fragments straight from pre-packed global (L2-resident weights)
        v16bf bfr[2];
#pragma unroll
        for (int ni = 0; ni < 2; ++ni) {
            const int nt = blockIdx.x * 8 + wn * 2 + ni;
            const unsigned short* bp = Bp + (((size_t)kt * ntileN + nt) * 32 + lane) * 16;
            bfr[ni] = ld_frag_pair(bp, bp + 8);
        }
#pragma unroll
        for (int mi = 0; mi < 4; ++mi)
#pragma unroll
            for (int ni = 0; ni < 2; ++ni)
                acc[mi][ni] = wmma_bf16(af[mi], bfr[ni], acc[mi][ni]);

        wait_async0();        // next tile's DMA has landed in the other buffer
        __syncthreads();
    }

    // epilogue: C layout -> VGPR r holds (row = r + 8*hf, col = ln)
#pragma unroll
    for (int mi = 0; mi < 4; ++mi) {
#pragma unroll
        for (int ni = 0; ni < 2; ++ni) {
            const int colg = blockN + wn * 32 + ni * 16 + ln;
            const float bv = bias[colg];
#pragma unroll
            for (int r = 0; r < 8; ++r) {
                const int rowg = blockM + wm * 64 + mi * 16 + r + 8 * hf;
                float v = acc[mi][ni][r] + bv;
                if (RESID) v += resid[(size_t)rowg * N + colg];
                if (RELU)  v = fmaxf(v, 0.0f);
                if (OUTF)  outF[(size_t)rowg * N + colg] = v;
                if (OUTB) {
                    size_t idx;
                    if (LAYOUT == 1) {        // [B,S,1024] -> [B,H,S,64]
                        const int bb = rowg >> 11, ss = rowg & 2047;
                        const int hh = colg >> 6,  dd = colg & 63;
                        idx = (((size_t)bb * 16 + hh) * 2048 + ss) * 64 + dd;
                    } else if (LAYOUT == 2) { // [B,S,1024] -> [B,H,64,S]  (V transposed)
                        const int bb = rowg >> 11, ss = rowg & 2047;
                        const int hh = colg >> 6,  dd = colg & 63;
                        idx = (((size_t)bb * 16 + hh) * 64 + dd) * 2048 + ss;
                    } else {
                        idx = (size_t)rowg * N + colg;
                    }
                    outB[idx] = f2bf(v);
                }
            }
        }
    }
}

// ---------------------------------------------------------------------------
// Flash attention, transposed orientation: S' = K @ Q^T (keys=rows, queries=cols).
//  - per-query softmax reductions are in-lane + ONE shfl_xor(16)
//  - softmax in base-2 scaled domain: v_pk_fma_f32 + raw v_exp_f32 per element
//  - O'^T = V^T @ P' : V^T is the A operand, staged with async DMA
//  - P' B-fragment built with an 8-op pairwise lane-16 exchange (no LDS)
// Q,K bf16 [B*H][S][64]; Vt bf16 [B*H][64][S]; out bf16 [B*S][1024].
// One block = one (b,h) x 128 queries (8 waves x 16 queries); K/Vt double-buffered.
// ---------------------------------------------------------------------------
__global__ __launch_bounds__(256) void attn_kernel(const unsigned short* __restrict__ Q,
                                                   const unsigned short* __restrict__ Km,
                                                   const unsigned short* __restrict__ Vt,
                                                   unsigned short* __restrict__ outB) {
    __shared__ alignas(16) unsigned short Klds[2][32 * 72];   // 32 keys x 64 d, pitch 72
    __shared__ alignas(16) unsigned short Vlds[2][64 * 40];   // 64 d x 32 keys, pitch 40

    const int tid  = threadIdx.x;
    const int lane = tid & 31;
    const int wid  = tid >> 5;
    const int hf   = lane >> 4;
    const int ln   = lane & 15;

    const int bh   = blockIdx.x >> 4;          // b*16 + h
    const int qblk = blockIdx.x & 15;
    const int q0   = qblk * 128 + wid * 16;    // this wave's 16 query columns

    // Q fragments as B operand: b[i] = Q[q0+ln][32c + 16h + i]  (32B contiguous)
    const unsigned short* qrow = Q + ((size_t)bh * 2048 + q0 + ln) * 64;
    v16bf qf[2];
    qf[0] = ld_frag_pair(qrow + 16 * hf,      qrow + 16 * hf + 8);
    qf[1] = ld_frag_pair(qrow + 32 + 16 * hf, qrow + 32 + 16 * hf + 8);

    // softmax scale folded with log2(e): p = 2^(s*SC2 - m2)
    const float SC2 = 0.125f * 1.44269504f;
    float m2 = -1e30f, l_i = 0.0f;             // per-query (per-lane) state, base-2 domain
    v8f o[4];
#pragma unroll
    for (int j = 0; j < 4; ++j)
#pragma unroll
        for (int r = 0; r < 8; ++r) o[j][r] = 0.0f;

    const int rowK = tid >> 3, segK = tid & 7; // K staging: 32 rows x 8x16B
    const int rowV = tid >> 2, segV = tid & 3; // Vt staging: 64 rows x 4x16B
    const unsigned short* kSrc = Km + (size_t)bh * 2048 * 64 + (size_t)rowK * 64 + segK * 8;
    const unsigned short* vSrc = Vt + (size_t)bh * 64 * 2048 + (size_t)rowV * 2048 + segV * 8;
    const unsigned kl0 = lds_addr(&Klds[0][rowK * 72 + segK * 8]);
    const unsigned kl1 = lds_addr(&Klds[1][rowK * 72 + segK * 8]);
    const unsigned vl0 = lds_addr(&Vlds[0][rowV * 40 + segV * 8]);
    const unsigned vl1 = lds_addr(&Vlds[1][rowV * 40 + segV * 8]);

    // preload tile 0 via async DMA
    async_copy_b128(kl0, kSrc);
    async_copy_b128(vl0, vSrc);
    wait_async0();
    __syncthreads();

    for (int kt = 0; kt < 64; ++kt) {          // 2048 keys / 32
        const int cur = kt & 1;
        const bool more = (kt + 1) < 64;
        if (more) {
            const size_t kk = (size_t)(kt + 1) * 32;
            async_copy_b128(cur ? kl0 : kl1, kSrc + kk * 64);
            async_copy_b128(cur ? vl0 : vl1, vSrc + kk);
        }

        // hoist ALL K fragments (8x ds_load_b128) before the 4 score WMMAs
        v16bf kf[2][2];
#pragma unroll
        for (int c = 0; c < 2; ++c) {
            const int b0 = ln * 72 + 32 * c;          // keys 0..15
            const int b1 = (16 + ln) * 72 + 32 * c;   // keys 16..31
            kf[0][c] = ld_frag_pair(&Klds[cur][b0 + 16 * hf], &Klds[cur][b0 + 16 * hf + 8]);
            kf[1][c] = ld_frag_pair(&Klds[cur][b1 + 16 * hf], &Klds[cur][b1 + 16 * hf + 8]);
        }
        // hoist all V^T fragments too (overlaps with score WMMAs below)
        v16bf vf[4];
#pragma unroll
        for (int j = 0; j < 4; ++j) {
            const int vb = (16 * j + ln) * 40;
            vf[j] = ld_frag_pair(&Vlds[cur][vb + 16 * hf], &Vlds[cur][vb + 16 * hf + 8])
        }

        // S' = K @ Q^T for 2 key-subtiles of 16 (A = K rows, B = Q^T)
        v8f s0, s1;
#pragma unroll
        for (int r = 0; r < 8; ++r) { s0[r] = 0.0f; s1[r] = 0.0f; }
        s0 = wmma_bf16(kf[0][0], qf[0], s0);
        s1 = wmma_bf16(kf[1][0], qf[0], s1);
        s0 = wmma_bf16(kf[0][1], qf[1], s0);
        s1 = wmma_bf16(kf[1][1], qf[1], s1);

        // online softmax over keys (columns of each lane): in-lane tree + one xor-16
        float t = fmaxf(s0[0], s1[0]);
#pragma unroll
        for (int r = 1; r < 8; ++r) t = fmaxf(t, fmaxf(s0[r], s1[r]));
        t *= SC2;
        t = fmaxf(t, __shfl_xor(t, 16, 32));
        const float mn   = fmaxf(m2, t);
        const float corr = fast_exp2(m2 - mn);
        float sum = 0.0f;
#pragma unroll
        for (int r = 0; r < 8; ++r) {
            s0[r] = fast_exp2(fmaf(s0[r], SC2, -mn));
            s1[r] = fast_exp2(fmaf(s1[r], SC2, -mn));
            sum += s0[r] + s1[r];
        }
        sum += __shfl_xor(sum, 16, 32);
        l_i = l_i * corr + sum;
        m2 = mn;
#pragma unroll
        for (int j = 0; j < 4; ++j)
#pragma unroll
            for (int r = 0; r < 8; ++r) o[j][r] *= corr;

        // P' B-fragment: b[i] = P'[key=16h+i][q=ln]; pairwise lane-16 exchange
        FragU pu;
#pragma unroll
        for (int r = 0; r < 8; ++r) {
            const float send = hf ? s0[r] : s1[r];
            const float recv = __shfl_xor(send, 16, 32);
            const float lo = hf ? recv : s0[r];
            const float hi = hf ? s1[r] : recv;
            pu.h[r]     = f2bf(lo);
            pu.h[8 + r] = f2bf(hi);
        }

        // O'[j] += Vt(subtile j) @ P'
        o[0] = wmma_bf16(vf[0], pu.v, o[0]);
        o[1] = wmma_bf16(vf[1], pu.v, o[1]);
        o[2] = wmma_bf16(vf[2], pu.v, o[2]);
        o[3] = wmma_bf16(vf[3], pu.v, o[3]);

        wait_async0();        // next tile's DMA has landed in the other buffer
        __syncthreads();
    }

    // epilogue: lane owns query q0+ln; d = 16j + 8hf + r is contiguous -> 16B stores
    const int b = bh >> 4, h = bh & 15;
    const float invl = 1.0f / l_i;
    unsigned short* orow = outB + ((size_t)(b * 2048 + q0 + ln)) * 1024 + h * 64;
#pragma unroll
    for (int j = 0; j < 4; ++j) {
        HalfU u;
#pragma unroll
        for (int r = 0; r < 8; ++r) u.h[r] = f2bf(o[j][r] * invl);
        *reinterpret_cast<uint4*>(orow + 16 * j + 8 * hf) = u.q;
    }
}

// ---------------------------------------------------------------------------
// LayerNorm over last dim D (=1024); one row per block.
// ---------------------------------------------------------------------------
__global__ __launch_bounds__(256) void ln_kernel(const float* __restrict__ in,
                                                 const float* __restrict__ g,
                                                 const float* __restrict__ b,
                                                 float* __restrict__ outF,
                                                 unsigned short* __restrict__ outB,
                                                 int D) {
    __shared__ float red[256];
    const int tid = threadIdx.x;
    const size_t base = (size_t)blockIdx.x * D;

    float s = 0.0f, s2 = 0.0f;
    for (int c = tid; c < D; c += 256) {
        const float v = in[base + c];
        s += v;
        s2 += v * v;
    }
    red[tid] = s;
    __syncthreads();
    for (int off = 128; off > 0; off >>= 1) {
        if (tid < off) red[tid] += red[tid + off];
        __syncthreads();
    }
    const float mean = red[0] / D;
    __syncthreads();
    red[tid] = s2;
    __syncthreads();
    for (int off = 128; off > 0; off >>= 1) {
        if (tid < off) red[tid] += red[tid + off];
        __syncthreads();
    }
    const float var = red[0] / D - mean * mean;
    const float rs  = rsqrtf(var + 1e-5f);

    for (int c = tid; c < D; c += 256) {
        const float v  = in[base + c];
        const float ov = (v - mean) * rs * g[c] + b[c];
        outF[base + c] = ov;
        if (outB) outB[base + c] = f2bf(ov);
    }
}

// ---------------------------------------------------------------------------
// launcher
// ---------------------------------------------------------------------------
extern "C" void kernel_launch(void* const* d_in, const int* in_sizes, int n_in,
                              void* d_out, int out_size, void* d_ws, size_t ws_size,
                              hipStream_t stream) {
    (void)in_sizes; (void)n_in; (void)out_size; (void)ws_size;

    const int M = 8192;      // B*S
    const int D = 1024;      // d_model
    const int F = 4096;      // d_ff
    const size_t MD = (size_t)M * D;
    const size_t MF = (size_t)M * F;

    const float* x   = (const float*)d_in[0];
    const float* Wq  = (const float*)d_in[1];
    const float* bq  = (const float*)d_in[2];
    const float* Wk  = (const float*)d_in[3];
    const float* bk  = (const float*)d_in[4];
    const float* Wv  = (const float*)d_in[5];
    const float* bv  = (const float*)d_in[6];
    const float* Wo  = (const float*)d_in[7];
    const float* bo  = (const float*)d_in[8];
    const float* W1  = (const float*)d_in[9];
    const float* b1  = (const float*)d_in[10];
    const float* W2  = (const float*)d_in[11];
    const float* b2  = (const float*)d_in[12];
    const float* g1  = (const float*)d_in[13];
    const float* be1 = (const float*)d_in[14];
    const float* g2  = (const float*)d_in[15];
    const float* be2 = (const float*)d_in[16];

    char* w = (char*)d_ws;
    size_t off = 0;
    auto take = [&](size_t bytes) -> void* {
        void* p = w + off;
        off = (off + bytes + 255) & ~(size_t)255;
        return p;
    };

    unsigned short* xb    = (unsigned short*)take(MD * 2);
    unsigned short* qb    = (unsigned short*)take(MD * 2);
    unsigned short* kb    = (unsigned short*)take(MD * 2);
    unsigned short* vtb   = (unsigned short*)take(MD * 2);   // V transposed [B,H,64,S]
    unsigned short* attnb = (unsigned short*)take(MD * 2);
    unsigned short* x1b   = (unsigned short*)take(MD * 2);
    unsigned short* hb    = (unsigned short*)take(MF * 2);
    unsigned short* wqp   = (unsigned short*)take((size_t)D * D * 2);
    unsigned short* wkp   = (unsigned short*)take((size_t)D * D * 2);
    unsigned short* wvp   = (unsigned short*)take((size_t)D * D * 2);
    unsigned short* wop   = (unsigned short*)take((size_t)D * D * 2);
    unsigned short* w1p   = (unsigned short*)take((size_t)D * F * 2);
    unsigned short* w2p   = (unsigned short*)take((size_t)F * D * 2);
    float*          r1    = (float*)take(MD * 4);
    float*          x1f   = (float*)take(MD * 4);
    float*          r2    = (float*)take(MD * 4);

    // 1) convert activations, pack weights to WMMA fragment layout
    cvt_bf16_kernel<<<32768, 256, 0, stream>>>(x, xb, MD);
    packw_kernel<<<(unsigned)((size_t)D * D / 256), 256, 0, stream>>>(Wq, wqp, D, D);
    packw_kernel<<<(unsigned)((size_t)D * D / 256), 256, 0, stream>>>(Wk, wkp, D, D);
    packw_kernel<<<(unsigned)((size_t)D * D / 256), 256, 0, stream>>>(Wv, wvp, D, D);
    packw_kernel<<<(unsigned)((size_t)D * D / 256), 256, 0, stream>>>(Wo, wop, D, D);
    packw_kernel<<<(unsigned)((size_t)D * F / 256), 256, 0, stream>>>(W1, w1p, D, F);
    packw_kernel<<<(unsigned)((size_t)F * D / 256), 256, 0, stream>>>(W2, w2p, F, D);

    // 2) QKV projections: Q,K -> [B,H,S,64]; V -> [B,H,64,S] (transposed for attention A-operand)
    dim3 gP(D / 128, M / 128);
    gemm_kernel<1, false, false, false, true><<<gP, 256, 0, stream>>>(
        xb, wqp, bq, nullptr, nullptr, qb, M, D, D);
    gemm_kernel<1, false, false, false, true><<<gP, 256, 0, stream>>>(
        xb, wkp, bk, nullptr, nullptr, kb, M, D, D);
    gemm_kernel<2, false, false, false, true><<<gP, 256, 0, stream>>>(
        xb, wvp, bv, nullptr, nullptr, vtb, M, D, D);

    // 3) flash attention: 64 (b,h) x 16 query-blocks
    attn_kernel<<<1024, 256, 0, stream>>>(qb, kb, vtb, attnb);

    // 4) output projection + residual -> r1 = x + attn@Wo + bo
    gemm_kernel<0, false, true, true, false><<<gP, 256, 0, stream>>>(
        attnb, wop, bo, x, r1, nullptr, M, D, D);

    // 5) LN1 -> x1 (f32 + bf16)
    ln_kernel<<<M, 256, 0, stream>>>(r1, g1, be1, x1f, x1b, D);

    // 6) FFN1 with ReLU -> hb bf16 [M, 4096]
    dim3 gF1(F / 128, M / 128);
    gemm_kernel<0, true, false, false, true><<<gF1, 256, 0, stream>>>(
        x1b, w1p, b1, nullptr, nullptr, hb, M, F, D);

    // 7) FFN2 + residual -> r2 = x1 + h@W2 + b2
    gemm_kernel<0, false, true, true, false><<<gP, 256, 0, stream>>>(
        hb, w2p, b2, x1f, r2, nullptr, M, D, F);

    // 8) LN2 -> final f32 output
    ln_kernel<<<M, 256, 0, stream>>>(r2, g2, be2, (float*)d_out, nullptr, D);
}
